// GraphSAGE_62612033241324
// MI455X (gfx1250) — matrix operand order repaired and checked by reading, not compiled, not run on hardware
//
#include <hip/hip_runtime.h>
#include <hip/hip_bf16.h>

// ---------------- problem constants (from reference) ----------------
constexpr long long NN   = 50000;   // nodes
constexpr int IN   = 128;
constexpr int HID  = 512;
constexpr int OUT  = 40;
constexpr int K1   = 2 * IN;        // 256  ([agg | x] concat)
constexpr int K2   = 2 * HID;       // 1024 ([agg2 | h1] concat)
constexpr int KT1  = K1 / 32;       // 8 k-tiles, layer 1
constexpr int KT2  = K2 / 32;       // 32 k-tiles, layer 2
constexpr int NT1  = HID / 16;      // 32 n-tiles, layer 1
constexpr int NT2  = 3;             // 48-wide padded (40 real), layer 2
constexpr long long MT = NN / 16;   // 3125 exact 16-row tiles

typedef __attribute__((ext_vector_type(16))) __bf16 v16bf;
typedef __attribute__((ext_vector_type(8)))  float  v8f;

union BF16x16 { v16bf v; unsigned short u[16]; };

__device__ __forceinline__ unsigned short f2bf(float f) {
  // round-to-nearest-even fp32 -> bf16
  unsigned int x = __float_as_uint(f);
  x += 0x7FFFu + ((x >> 16) & 1u);
  return (unsigned short)(x >> 16);
}

// ---------------- utility: zero fill ----------------
__global__ void zero_kernel(float* __restrict__ p, long long n) {
  long long i = (long long)blockIdx.x * blockDim.x + threadIdx.x;
  if (i < n) p[i] = 0.0f;
}

// ---------------- pack W1 = [w1_l ; w1_r] into per-lane bf16 B fragments ----------------
// B fragment layout (16-bit B, 32x16): lane = (khalf<<4) | n_local ;
// element (v*2+t) holds K = (v>>2)*16 + khalf*8 + (v&3)*2 + t
__global__ void cvtw1_kernel(const float* __restrict__ w1_l,
                             const float* __restrict__ w1_r,
                             unsigned short* __restrict__ Wc1) {
  int t = blockIdx.x * blockDim.x + threadIdx.x;
  if (t >= KT1 * NT1 * 32) return;
  int lane = t & 31;
  int nt   = (t >> 5) % NT1;
  int kt   = t / (32 * NT1);
  int n     = nt * 16 + (lane & 15);
  int khalf = lane >> 4;
  unsigned short* dp = Wc1 + ((size_t)(kt * NT1 + nt) * 32 + lane) * 16;
  for (int v = 0; v < 8; ++v)
    for (int tt = 0; tt < 2; ++tt) {
      int k = kt * 32 + (v >> 2) * 16 + khalf * 8 + (v & 3) * 2 + tt;
      float val = (k < IN) ? w1_l[(size_t)k * HID + n]
                           : w1_r[(size_t)(k - IN) * HID + n];
      dp[v * 2 + tt] = f2bf(val);
    }
}

// ---------------- pack W2 = [w2_l ; w2_r] (N padded 40 -> 48) ----------------
__global__ void cvtw2_kernel(const float* __restrict__ w2_l,
                             const float* __restrict__ w2_r,
                             unsigned short* __restrict__ Wc2) {
  int t = blockIdx.x * blockDim.x + threadIdx.x;
  if (t >= KT2 * NT2 * 32) return;
  int lane = t & 31;
  int nt   = (t >> 5) % NT2;
  int kt   = t / (32 * NT2);
  int n     = nt * 16 + (lane & 15);
  int khalf = lane >> 4;
  unsigned short* dp = Wc2 + ((size_t)(kt * NT2 + nt) * 32 + lane) * 16;
  for (int v = 0; v < 8; ++v)
    for (int tt = 0; tt < 2; ++tt) {
      int k = kt * 32 + (v >> 2) * 16 + khalf * 8 + (v & 3) * 2 + tt;
      float val = 0.0f;
      if (n < OUT)
        val = (k < HID) ? w2_l[(size_t)k * OUT + n]
                        : w2_r[(size_t)(k - HID) * OUT + n];
      dp[v * 2 + tt] = f2bf(val);
    }
}

// ---------------- edge scatter, layer 1: one wave per edge (row of 128) ----------------
__global__ void scatter1_kernel(const float* __restrict__ x,
                                const int* __restrict__ src,
                                const int* __restrict__ dst,
                                float* __restrict__ agg1,
                                float* __restrict__ deg, int E) {
  long long g = (long long)blockIdx.x * blockDim.x + threadIdx.x;
  int e = (int)(g >> 5);
  int lane = (int)(g & 31);
  if (e >= E) return;
  int s = src[e], d = dst[e];
  const float4 v = *(const float4*)(x + (long long)s * IN + lane * 4);
  float* ap = agg1 + (long long)d * IN + lane * 4;
  atomicAdd(ap + 0, v.x);
  atomicAdd(ap + 1, v.y);
  atomicAdd(ap + 2, v.z);
  atomicAdd(ap + 3, v.w);
  if (lane == 0) atomicAdd(deg + d, 1.0f);
}

// ---------------- edge scatter, layer 2: 128 threads per edge (row of 512) ----------------
__global__ void scatter2_kernel(const float* __restrict__ h1,
                                const int* __restrict__ src,
                                const int* __restrict__ dst,
                                float* __restrict__ agg2, int E) {
  long long g = (long long)blockIdx.x * blockDim.x + threadIdx.x;
  int e = (int)(g >> 7);
  int r = (int)(g & 127);
  if (e >= E) return;
  int s = src[e], d = dst[e];
  const float4 v = *(const float4*)(h1 + (long long)s * HID + r * 4);
  float* ap = agg2 + (long long)d * HID + r * 4;
  atomicAdd(ap + 0, v.x);
  atomicAdd(ap + 1, v.y);
  atomicAdd(ap + 2, v.z);
  atomicAdd(ap + 3, v.w);
}

// ---------------- GEMM1 fused: h1 = relu([agg1/deg | x] @ W1 + b1) ----------------
// Block = 256 thr = 8 waves. Block owns one 16-row M tile; wave w owns N cols [w*64, w*64+64).
// A tile (16x32 bf16) staged in LDS in exact WMMA A-fragment order, shared by all 8 waves.
__global__ void __launch_bounds__(256) gemm1_kernel(
    const float* __restrict__ x, const float* __restrict__ agg1,
    const float* __restrict__ deg, const unsigned short* __restrict__ Wc1,
    const float* __restrict__ bias1, float* __restrict__ h1) {
  __shared__ __align__(32) unsigned short lds_a[32 * 16];

  const int tid  = threadIdx.x;
  const int lane = tid & 31;
  const int w    = tid >> 5;
  const long long m_base = (long long)blockIdx.x * 16;

  // cooperative-stage indices: thread -> (row m_local, k pair)
  const int ml      = tid >> 4;        // 0..15
  const int kp      = tid & 15;        // 0..15
  const int k_local = kp * 2;
  const int khalf_s = (k_local >> 3) & 1;
  const int v_s     = (k_local >> 4) * 4 + ((k_local >> 1) & 3);
  const int lds_idx = (ml + khalf_s * 16) * 16 + v_s * 2;
  const long long m_ld = m_base + ml;
  const float rd = 1.0f / fmaxf(deg[m_ld], 1.0f);

  v8f acc[4] = {};

  for (int kt = 0; kt < KT1; ++kt) {
    const int k = kt * 32 + k_local;
    float v0, v1;
    if (k < IN) {                       // aggregated-neighbor half, mean-scaled
      v0 = agg1[m_ld * IN + k]     * rd;
      v1 = agg1[m_ld * IN + k + 1] * rd;
    } else {                            // self half
      v0 = x[m_ld * IN + (k - IN)];
      v1 = x[m_ld * IN + (k - IN) + 1];
    }
    lds_a[lds_idx]     = f2bf(v0);
    lds_a[lds_idx + 1] = f2bf(v1);
    __syncthreads();

    v16bf a = *reinterpret_cast<const v16bf*>(lds_a + lane * 16);
    __syncthreads();

    const unsigned short* wb =
        Wc1 + ((size_t)(kt * NT1 + w * 4) * 32 + lane) * 16;
#pragma unroll
    for (int j = 0; j < 4; ++j) {
      v16bf b = *reinterpret_cast<const v16bf*>(wb + (size_t)j * 32 * 16);
      acc[j] = __builtin_amdgcn_wmma_f32_16x16x32_bf16(
          false, a, false, b, (short)0, acc[j], false, false);
    }
  }

  // epilogue: +bias, ReLU, store (C/D layout: lane n = lane&15, rows r + 8*(lane>>4))
  const int nl    = lane & 15;
  const int rbase = (lane >> 4) * 8;
#pragma unroll
  for (int j = 0; j < 4; ++j) {
    const int n = (w * 4 + j) * 16 + nl;
    const float bb = bias1[n];
#pragma unroll
    for (int r = 0; r < 8; ++r) {
      const long long m = m_base + rbase + r;
      h1[m * HID + n] = fmaxf(acc[j][r] + bb, 0.0f);
    }
  }
}

// ---------------- GEMM2 fused: logits = [agg2/deg | h1] @ W2 + b2 ----------------
// Wave-independent: each wave owns one 16-row M tile and all 48(40) output cols.
__global__ void __launch_bounds__(256) gemm2_kernel(
    const float* __restrict__ h1, const float* __restrict__ agg2,
    const float* __restrict__ deg, const unsigned short* __restrict__ Wc2,
    const float* __restrict__ bias2, float* __restrict__ out) {
  const int tid  = threadIdx.x;
  const int lane = tid & 31;
  const int w    = tid >> 5;
  const long long m_tile = (long long)blockIdx.x * 8 + w;
  if (m_tile >= MT) return;            // no barriers below -> safe

  const long long m = m_tile * 16 + (lane & 15);
  const int khalf   = lane >> 4;
  const float rd = 1.0f / fmaxf(deg[m], 1.0f);

  v8f acc[3] = {};

  for (int kt = 0; kt < KT2; ++kt) {
    BF16x16 au;
#pragma unroll
    for (int v = 0; v < 8; ++v) {
      // A fragment: element (v*2+t) holds K = (v>>2)*16 + khalf*8 + (v&3)*2 + t
      const int k = kt * 32 + (v >> 2) * 16 + khalf * 8 + (v & 3) * 2;
      float f0, f1;
      if (k < HID) {
        f0 = agg2[m * HID + k]     * rd;
        f1 = agg2[m * HID + k + 1] * rd;
      } else {
        f0 = h1[m * HID + (k - HID)];
        f1 = h1[m * HID + (k - HID) + 1];
      }
      au.u[v * 2]     = f2bf(f0);
      au.u[v * 2 + 1] = f2bf(f1);
    }
    const unsigned short* wb = Wc2 + ((size_t)(kt * NT2) * 32 + lane) * 16;
#pragma unroll
    for (int j = 0; j < 3; ++j) {
      v16bf b = *reinterpret_cast<const v16bf*>(wb + (size_t)j * 32 * 16);
      acc[j] = __builtin_amdgcn_wmma_f32_16x16x32_bf16(
          false, au.v, false, b, (short)0, acc[j], false, false);
    }
  }

  const int nl = lane & 15;
  const long long mrow0 = m_tile * 16 + khalf * 8;
#pragma unroll
  for (int j = 0; j < 3; ++j) {
    const int n = j * 16 + nl;
    if (n < OUT) {
      const float bb = bias2[n];
#pragma unroll
      for (int r = 0; r < 8; ++r)
        out[(mrow0 + r) * OUT + n] = acc[j][r] + bb;
    }
  }
}

// ---------------- in-place log_softmax over 40 classes ----------------
__global__ void lsm_kernel(float* __restrict__ out) {
  long long i = (long long)blockIdx.x * blockDim.x + threadIdx.x;
  if (i >= NN) return;
  float* row = out + i * OUT;
  float mx = row[0];
#pragma unroll
  for (int j = 1; j < OUT; ++j) mx = fmaxf(mx, row[j]);
  float s = 0.0f;
#pragma unroll
  for (int j = 0; j < OUT; ++j) s += expf(row[j] - mx);
  const float lse = mx + logf(s);
#pragma unroll
  for (int j = 0; j < OUT; ++j) row[j] -= lse;
}

// ---------------- host ----------------
extern "C" void kernel_launch(void* const* d_in, const int* in_sizes, int n_in,
                              void* d_out, int out_size, void* d_ws, size_t ws_size,
                              hipStream_t stream) {
  const float* x    = (const float*)d_in[0];
  const int*   edge = (const int*)d_in[1];
  const float* w1_l = (const float*)d_in[2];
  const float* b1   = (const float*)d_in[3];
  const float* w1_r = (const float*)d_in[4];
  const float* w2_l = (const float*)d_in[5];
  const float* b2   = (const float*)d_in[6];
  const float* w2_r = (const float*)d_in[7];
  const int E = in_sizes[1] / 2;
  const int* src = edge;
  const int* dst = edge + E;
  float* out = (float*)d_out;

  // workspace carve-up (256B aligned)
  char* ws = (char*)d_ws;
  size_t off = 0;
  auto take = [&](size_t bytes) -> char* {
    char* p = ws + off;
    off = (off + bytes + 255) & ~(size_t)255;
    return p;
  };
  float* deg  = (float*)take(NN * 4);
  float* agg1 = (float*)take(NN * (size_t)IN * 4);
  float* h1   = (float*)take(NN * (size_t)HID * 4);
  float* agg2 = (float*)take(NN * (size_t)HID * 4);
  unsigned short* Wc1 = (unsigned short*)take((size_t)KT1 * NT1 * 32 * 16 * 2);
  unsigned short* Wc2 = (unsigned short*)take((size_t)KT2 * NT2 * 32 * 16 * 2);

  const int B = 256;
  auto blks = [](long long n, int b) { return (unsigned)((n + b - 1) / b); };

  // zero accumulation targets (every call: deterministic)
  zero_kernel<<<blks(NN, B), B, 0, stream>>>(deg, NN);
  zero_kernel<<<blks(NN * IN, B), B, 0, stream>>>(agg1, NN * IN);
  zero_kernel<<<blks(NN * HID, B), B, 0, stream>>>(agg2, NN * (long long)HID);

  // pack weights to bf16 WMMA fragments
  cvtw1_kernel<<<blks((long long)KT1 * NT1 * 32, B), B, 0, stream>>>(w1_l, w1_r, Wc1);
  cvtw2_kernel<<<blks((long long)KT2 * NT2 * 32, B), B, 0, stream>>>(w2_l, w2_r, Wc2);

  // layer 1: aggregate + fused GEMM
  scatter1_kernel<<<blks((long long)E * 32, B), B, 0, stream>>>(x, src, dst, agg1, deg, E);
  gemm1_kernel<<<(unsigned)MT, B, 0, stream>>>(x, agg1, deg, Wc1, b1, h1);

  // layer 2: aggregate + fused GEMM
  scatter2_kernel<<<blks((long long)E * 128, B), B, 0, stream>>>(h1, src, dst, agg2, E);
  gemm2_kernel<<<blks(MT, 8), B, 0, stream>>>(h1, agg2, deg, Wc2, b2, out);

  // log_softmax in place on d_out
  lsm_kernel<<<blks(NN, B), B, 0, stream>>>(out);
}
